// HybridGCNGAT_2516850835930
// MI455X (gfx1250) — matrix operand order repaired and checked by reading, not compiled, not run on hardware
//
#include <hip/hip_runtime.h>
#include <hip/hip_bf16.h>
#include <math.h>

// Model dims (fixed by the reference)
#define IN_CH   128
#define HID     64
#define OUTC    128
#define NHEAD   4
#define GAT_HID (NHEAD * HID)   // 256
#define HOUT    (NHEAD * OUTC)  // 512

typedef __attribute__((ext_vector_type(16))) __bf16          v16bf;
typedef __attribute__((ext_vector_type(16))) unsigned short  v16us;
typedef __attribute__((ext_vector_type(8)))  unsigned short  v8us;
typedef __attribute__((ext_vector_type(8)))  float           v8f;

// ---------------- helpers ----------------
__device__ __forceinline__ unsigned short f2bf(float f) {
  unsigned u = __float_as_uint(f);
  u += 0x7FFFu + ((u >> 16) & 1u);     // round-to-nearest-even
  return (unsigned short)(u >> 16);
}
__device__ __forceinline__ float lrelu(float x) { return x > 0.f ? x : 0.2f * x; }
// order-preserving float<->uint encoding for atomicMax on floats
__device__ __forceinline__ unsigned fenc(float f) {
  unsigned u = __float_as_uint(f);
  return (u & 0x80000000u) ? ~u : (u | 0x80000000u);
}
__device__ __forceinline__ float fdec(unsigned u) {
  u = (u & 0x80000000u) ? (u & 0x7FFFFFFFu) : ~u;
  return __uint_as_float(u);
}

// ---------------- WMMA bf16 GEMM ----------------
// C[M,NC] = A[M,K]*B[K,NC] (+bias).  A: row-major bf16.  Bp: pre-packed bf16 in
// WMMA B-fragment order: [(tileN*(K/32)+kstep)*32 + lane] * 16 elems (32B/lane).
// One wave per 16x16 C tile; K is a template constant so the k-loop fully unrolls
// into  (2 b128 A-loads + 2 b128 B-loads + 1 v_wmma)  per step.
template <int K>
__global__ void k_gemm_bf16(const unsigned short* __restrict__ A,
                            const unsigned short* __restrict__ Bp,
                            const float* __restrict__ bias, float* __restrict__ C,
                            int M, int NC) {
  const int lane = threadIdx.x & 31;
  const int h = lane >> 4;     // K-half selector
  const int p = lane & 15;     // A-row / C-col within tile
  const int tm = blockIdx.y << 4;
  const int tn = blockIdx.x << 4;
  constexpr int KSTEPS = K >> 5;

  v8f acc = {0.f, 0.f, 0.f, 0.f, 0.f, 0.f, 0.f, 0.f};
  const unsigned short* arow = A + (size_t)(tm + p) * K + h * 8;
  const unsigned short* brow = Bp + ((size_t)blockIdx.x * KSTEPS * 32 + lane) * 16;

#pragma unroll
  for (int ks = 0; ks < KSTEPS; ++ks) {
    v8us alo = *(const v8us*)(arow + ks * 32);        // K = h*8 + 0..7
    v8us ahi = *(const v8us*)(arow + ks * 32 + 16);   // K = 16 + h*8 + 0..7
    v8us blo = *(const v8us*)(brow + ks * 512);
    v8us bhi = *(const v8us*)(brow + ks * 512 + 8);
    v16us au = __builtin_shufflevector(alo, ahi, 0,1,2,3,4,5,6,7,8,9,10,11,12,13,14,15);
    v16us bu = __builtin_shufflevector(blo, bhi, 0,1,2,3,4,5,6,7,8,9,10,11,12,13,14,15);
    acc = __builtin_amdgcn_wmma_f32_16x16x32_bf16(false, __builtin_bit_cast(v16bf, au),
                                                  false, __builtin_bit_cast(v16bf, bu),
                                                  (short)0, acc, false, false);
  }
  float bb = bias ? bias[tn + p] : 0.f;
#pragma unroll
  for (int r = 0; r < 8; ++r) {                       // C: row = h*8+r, col = p
    C[(size_t)(tm + h * 8 + r) * NC + tn + p] = acc[r] + bb;
  }
}

// pack f32 weight B[K,NC] -> bf16 WMMA fragment order (tiny, runs once per launch)
__global__ void k_repack_w(const float* __restrict__ B, unsigned short* __restrict__ Bp,
                           int K, int NC) {
  int idx = blockIdx.x * blockDim.x + threadIdx.x;
  int ksteps = K >> 5;
  int total = (NC >> 4) * ksteps * 32;
  if (idx >= total) return;
  int lane = idx & 31;
  int ks = (idx >> 5) % ksteps;
  int tn = (idx >> 5) / ksteps;
  int h = lane >> 4, nn = (tn << 4) + (lane & 15);
  unsigned short* o = Bp + (size_t)idx * 16;
#pragma unroll
  for (int t = 0; t < 16; ++t)
    o[t] = f2bf(B[(size_t)(ks * 32 + h * 8 + t + (t >= 8 ? 8 : 0)) * NC + nn]);
}
__global__ void k_f32_to_bf16(const float* __restrict__ in, unsigned short* __restrict__ out,
                              size_t cnt) {
  size_t i = (size_t)blockIdx.x * blockDim.x + threadIdx.x;
  if (i < cnt) out[i] = f2bf(in[i]);
}

// ---------------- graph / elementwise kernels ----------------
__global__ void k_fill(float* p, float v, int n) {
  int i = blockIdx.x * blockDim.x + threadIdx.x;
  if (i < n) p[i] = v;
}
__global__ void k_deg_edge(const int* __restrict__ dst, float* deg, int E) {
  int e = blockIdx.x * blockDim.x + threadIdx.x;
  if (e < E) atomicAdd(&deg[dst[e]], 1.0f);
}
__global__ void k_rsqrt(float* deg, int n) {
  int i = blockIdx.x * blockDim.x + threadIdx.x;
  if (i < n) deg[i] = rsqrtf(deg[i]);          // deg >= 1 (self loop)
}
__global__ void k_gcn_init(const float* __restrict__ xw, const float* __restrict__ dis,
                           float* __restrict__ out, int n, int C) {
  size_t idx = (size_t)blockIdx.x * blockDim.x + threadIdx.x;
  if (idx >= (size_t)n * C) return;
  int i = idx / C;
  out[idx] = dis[i] * dis[i] * xw[idx];
}
__global__ void k_gcn_edge(const int* __restrict__ src, const int* __restrict__ dst,
                           const float* __restrict__ xw, const float* __restrict__ dis,
                           float* __restrict__ out, int E, int C) {
  size_t idx = (size_t)blockIdx.x * blockDim.x + threadIdx.x;
  if (idx >= (size_t)E * C) return;
  int e = idx / C, c = idx % C;
  int s = src[e], d = dst[e];
  atomicAdd(&out[(size_t)d * C + c], dis[s] * dis[d] * xw[(size_t)s * C + c]);
}
__global__ void k_bias_act(float* __restrict__ x, const float* __restrict__ b,
                           int n, int C, int relu) {
  size_t idx = (size_t)blockIdx.x * blockDim.x + threadIdx.x;
  if (idx >= (size_t)n * C) return;
  float v = x[idx] + b[idx % C];
  x[idx] = relu ? fmaxf(v, 0.f) : v;
}
// per-(node,head) attention logits: one wave per row of C elems
__global__ void k_gat_alpha(const float* __restrict__ xh, const float* __restrict__ asw,
                            const float* __restrict__ adw, float* __restrict__ as_,
                            float* __restrict__ ad_, int n, int Hh, int C) {
  int wid  = (blockIdx.x * blockDim.x + threadIdx.x) >> 5;
  int lane = threadIdx.x & 31;
  if (wid >= n * Hh) return;
  int i = wid / Hh, hh = wid % Hh;
  const float* row = xh + ((size_t)i * Hh + hh) * C;
  float ss = 0.f, dd = 0.f;
  for (int c = lane; c < C; c += 32) {
    float v = row[c];
    ss += v * asw[hh * C + c];
    dd += v * adw[hh * C + c];
  }
  for (int o = 16; o > 0; o >>= 1) { ss += __shfl_xor(ss, o, 32); dd += __shfl_xor(dd, o, 32); }
  if (lane == 0) { as_[i * Hh + hh] = ss; ad_[i * Hh + hh] = dd; }
}
__global__ void k_gat_max_init(const float* as_, const float* ad_, unsigned* m, int nH) {
  int i = blockIdx.x * blockDim.x + threadIdx.x;
  if (i < nH) m[i] = fenc(lrelu(as_[i] + ad_[i]));        // self loop term
}
__global__ void k_gat_max_edge(const int* __restrict__ src, const int* __restrict__ dst,
                               const float* __restrict__ as_, const float* __restrict__ ad_,
                               unsigned* __restrict__ m, int E, int Hh) {
  int idx = blockIdx.x * blockDim.x + threadIdx.x;
  if (idx >= E * Hh) return;
  int e = idx / Hh, hh = idx % Hh;
  int s = src[e], d = dst[e];
  atomicMax(&m[d * Hh + hh], fenc(lrelu(as_[s * Hh + hh] + ad_[d * Hh + hh])));
}
__global__ void k_gat_max_dec(unsigned* m, int nH) {
  int i = blockIdx.x * blockDim.x + threadIdx.x;
  if (i < nH) { float f = fdec(m[i]); ((float*)m)[i] = f; }
}
__global__ void k_gat_den_init(const float* as_, const float* ad_, const float* m,
                               float* den, int nH) {
  int i = blockIdx.x * blockDim.x + threadIdx.x;
  if (i < nH) den[i] = __expf(lrelu(as_[i] + ad_[i]) - m[i]);   // self loop
}
__global__ void k_gat_out_init(const float* __restrict__ xh, const float* __restrict__ den,
                               float* __restrict__ out, int n, int Hh, int C) {
  size_t idx = (size_t)blockIdx.x * blockDim.x + threadIdx.x;
  if (idx >= (size_t)n * Hh * C) return;
  int i = idx / (Hh * C);
  int hh = (idx / C) % Hh;
  out[idx] = den[i * Hh + hh] * xh[idx];
}
// one block per edge; blockDim = Hh*C (256 or 512)
__global__ void k_gat_edge(const int* __restrict__ src, const int* __restrict__ dst,
                           const float* __restrict__ as_, const float* __restrict__ ad_,
                           const float* __restrict__ m, const float* __restrict__ xh,
                           float* __restrict__ out, float* __restrict__ den, int Hh, int C) {
  int e = blockIdx.x;
  int t = threadIdx.x;
  int hh = t / C, c = t % C;
  int s = src[e], d = dst[e];
  float ev = lrelu(as_[s * Hh + hh] + ad_[d * Hh + hh]);
  float ee = __expf(ev - m[d * Hh + hh]);
  atomicAdd(&out[((size_t)d * Hh + hh) * C + c], ee * xh[((size_t)s * Hh + hh) * C + c]);
  if (c == 0) atomicAdd(&den[d * Hh + hh], ee);
}
// layer-1 finalize: g1 = elu(out/den + b)   (in place)
__global__ void k_gat_fin1(float* __restrict__ out, const float* __restrict__ den,
                           const float* __restrict__ b, int n, int Hh, int C) {
  size_t idx = (size_t)blockIdx.x * blockDim.x + threadIdx.x;
  if (idx >= (size_t)n * Hh * C) return;
  int i = idx / (Hh * C);
  int j = idx % (Hh * C);
  int hh = j / C;
  float v = out[idx] / den[i * Hh + hh] + b[j];
  out[idx] = v > 0.f ? v : expm1f(v);
}
// layer-2 finalize: x_gat[i,c] = mean_h(out/den) + b
__global__ void k_gat_fin2(const float* __restrict__ out, const float* __restrict__ den,
                           const float* __restrict__ b, float* __restrict__ xg,
                           int n, int Hh, int C) {
  size_t idx = (size_t)blockIdx.x * blockDim.x + threadIdx.x;
  if (idx >= (size_t)n * C) return;
  int i = idx / C, c = idx % C;
  float s = 0.f;
  for (int hh = 0; hh < Hh; ++hh)
    s += out[((size_t)i * Hh + hh) * C + c] / den[i * Hh + hh];
  xg[idx] = s / Hh + b[c];
}
// LayerNorm one wave per row, write bf16 into the concat buffer
__global__ void k_layernorm_bf16(const float* __restrict__ x, const float* __restrict__ g,
                                 const float* __restrict__ b, unsigned short* __restrict__ dst,
                                 int n, int C, int dstStride, int dstOff) {
  int wid  = (blockIdx.x * blockDim.x + threadIdx.x) >> 5;
  int lane = threadIdx.x & 31;
  if (wid >= n) return;
  const float* row = x + (size_t)wid * C;
  float s = 0.f;
  for (int c = lane; c < C; c += 32) s += row[c];
  for (int o = 16; o > 0; o >>= 1) s += __shfl_xor(s, o, 32);
  float mu = s / C;
  float v = 0.f;
  for (int c = lane; c < C; c += 32) { float d = row[c] - mu; v += d * d; }
  for (int o = 16; o > 0; o >>= 1) v += __shfl_xor(v, o, 32);
  float rs = rsqrtf(v / C + 1e-5f);
  unsigned short* drow = dst + (size_t)wid * dstStride + dstOff;
  for (int c = lane; c < C; c += 32) drow[c] = f2bf((row[c] - mu) * rs * g[c] + b[c]);
}

// ---------------- launch ----------------
extern "C" void kernel_launch(void* const* d_in, const int* in_sizes, int n_in,
                              void* d_out, int out_size, void* d_ws, size_t ws_size,
                              hipStream_t stream) {
  const float* x      = (const float*)d_in[0];
  const int*   ei     = (const int*)d_in[1];
  const float* W_gcn1 = (const float*)d_in[2];
  const float* b_gcn1 = (const float*)d_in[3];
  const float* W_gcn2 = (const float*)d_in[4];
  const float* b_gcn2 = (const float*)d_in[5];
  const float* W_gat1 = (const float*)d_in[6];
  const float* a_src1 = (const float*)d_in[7];
  const float* a_dst1 = (const float*)d_in[8];
  const float* b_gat1 = (const float*)d_in[9];
  const float* W_gat2 = (const float*)d_in[10];
  const float* a_src2 = (const float*)d_in[11];
  const float* a_dst2 = (const float*)d_in[12];
  const float* b_gat2 = (const float*)d_in[13];
  const float* g_gcn  = (const float*)d_in[14];
  const float* bt_gcn = (const float*)d_in[15];
  const float* g_gat  = (const float*)d_in[16];
  const float* bt_gat = (const float*)d_in[17];
  const float* W_fuse = (const float*)d_in[18];
  const float* b_fuse = (const float*)d_in[19];

  const int n = in_sizes[0] / IN_CH;   // 50000 (divisible by 16)
  const int E = in_sizes[1] / 2;       // 400000
  const int* src = ei;
  const int* dst = ei + E;

  // ---- workspace layout (~275 MB peak, big region time-shared) ----
  char* ws = (char*)d_ws;
  auto AL = [](size_t v) { return (v + 255) & ~(size_t)255; };
  size_t cur = 0;
  auto alloc = [&](size_t bytes) { size_t o = cur; cur = AL(cur + bytes); return o; };
  size_t o_deg  = alloc((size_t)n * 4);
  size_t o_xgcn = alloc((size_t)n * OUTC * 4);
  size_t o_xgat = alloc((size_t)n * OUTC * 4);
  size_t o_as   = alloc((size_t)n * NHEAD * 4);
  size_t o_ad   = alloc((size_t)n * NHEAD * 4);
  size_t o_m    = alloc((size_t)n * NHEAD * 4);
  size_t o_den  = alloc((size_t)n * NHEAD * 4);
  size_t o_xbf  = alloc((size_t)n * IN_CH * 2);          // x in bf16 (GEMM1, GEMM3)
  size_t o_wp1  = alloc((size_t)IN_CH * HID * 2);        // packed weights (bf16)
  size_t o_wp2  = alloc((size_t)HID * OUTC * 2);
  size_t o_wp3  = alloc((size_t)IN_CH * GAT_HID * 2);
  size_t o_wp4  = alloc((size_t)GAT_HID * HOUT * 2);
  size_t o_wp5  = alloc((size_t)(2 * OUTC) * OUTC * 2);
  size_t o_big  = cur;
  // phase A (GCN): xw1 | agg1 | xw2 ; h_bf reuses dead xw1 slot
  size_t b_xw1  = o_big;
  size_t b_agg1 = AL(b_xw1 + (size_t)n * HID * 4);
  size_t b_xw2  = AL(b_agg1 + (size_t)n * HID * 4);
  size_t b_hbf  = o_big;                                 // n*HID*2, after xw1 dead
  // phase B (GAT1): xh1 | out1 | xh2(high).  g1_bf reuses dead xh1 slot.
  size_t b_xh1  = o_big;
  size_t b_out1 = AL(b_xh1 + (size_t)n * GAT_HID * 4);
  size_t b_xh2  = AL(b_out1 + (size_t)n * GAT_HID * 4);
  size_t b_g1bf = o_big;                                 // n*GAT_HID*2, after xh1 dead
  size_t b_out2 = o_big;                                 // reuses g1_bf+out1 after GEMM4
  size_t b_cat  = o_big;                                 // bf16 concat, after out2 dead

  float* deg  = (float*)(ws + o_deg);
  float* xgcn = (float*)(ws + o_xgcn);
  float* xgat = (float*)(ws + o_xgat);
  float* as_  = (float*)(ws + o_as);
  float* ad_  = (float*)(ws + o_ad);
  float* mF   = (float*)(ws + o_m);
  unsigned* mU = (unsigned*)(ws + o_m);
  float* den  = (float*)(ws + o_den);
  unsigned short* xbf  = (unsigned short*)(ws + o_xbf);
  unsigned short* wp1  = (unsigned short*)(ws + o_wp1);
  unsigned short* wp2  = (unsigned short*)(ws + o_wp2);
  unsigned short* wp3  = (unsigned short*)(ws + o_wp3);
  unsigned short* wp4  = (unsigned short*)(ws + o_wp4);
  unsigned short* wp5  = (unsigned short*)(ws + o_wp5);
  float* xw1  = (float*)(ws + b_xw1);
  float* agg1 = (float*)(ws + b_agg1);
  float* xw2  = (float*)(ws + b_xw2);
  unsigned short* hbf = (unsigned short*)(ws + b_hbf);
  float* xh1  = (float*)(ws + b_xh1);
  float* out1 = (float*)(ws + b_out1);
  float* xh2  = (float*)(ws + b_xh2);
  unsigned short* g1bf = (unsigned short*)(ws + b_g1bf);
  float* out2 = (float*)(ws + b_out2);
  unsigned short* cat = (unsigned short*)(ws + b_cat);

  const int T = 256;
  auto blk = [&](size_t work) { return (unsigned)((work + T - 1) / T); };
  const int nH = n * NHEAD;

  // P0: pack weights (tiny) + x -> bf16
  k_repack_w<<<blk(IN_CH * HID / 16), T, 0, stream>>>(W_gcn1, wp1, IN_CH, HID);
  k_repack_w<<<blk(HID * OUTC / 16), T, 0, stream>>>(W_gcn2, wp2, HID, OUTC);
  k_repack_w<<<blk(IN_CH * GAT_HID / 16), T, 0, stream>>>(W_gat1, wp3, IN_CH, GAT_HID);
  k_repack_w<<<blk(GAT_HID * HOUT / 16), T, 0, stream>>>(W_gat2, wp4, GAT_HID, HOUT);
  k_repack_w<<<blk(2 * OUTC * OUTC / 16), T, 0, stream>>>(W_fuse, wp5, 2 * OUTC, OUTC);
  k_f32_to_bf16<<<blk((size_t)n * IN_CH), T, 0, stream>>>(x, xbf, (size_t)n * IN_CH);

  // P1: symmetric-normalized degrees (with self loops)
  k_fill<<<blk(n), T, 0, stream>>>(deg, 1.0f, n);
  k_deg_edge<<<blk(E), T, 0, stream>>>(dst, deg, E);
  k_rsqrt<<<blk(n), T, 0, stream>>>(deg, n);

  // P2-P3: GCN layer 1
  k_gemm_bf16<IN_CH><<<dim3(HID / 16, n / 16), 32, 0, stream>>>(xbf, wp1, nullptr, xw1, n, HID);
  k_gcn_init<<<blk((size_t)n * HID), T, 0, stream>>>(xw1, deg, agg1, n, HID);
  k_gcn_edge<<<blk((size_t)E * HID), T, 0, stream>>>(src, dst, xw1, deg, agg1, E, HID);
  k_bias_act<<<blk((size_t)n * HID), T, 0, stream>>>(agg1, b_gcn1, n, HID, 1);
  k_f32_to_bf16<<<blk((size_t)n * HID), T, 0, stream>>>(agg1, hbf, (size_t)n * HID);

  // P4-P5: GCN layer 2
  k_gemm_bf16<HID><<<dim3(OUTC / 16, n / 16), 32, 0, stream>>>(hbf, wp2, nullptr, xw2, n, OUTC);
  k_gcn_init<<<blk((size_t)n * OUTC), T, 0, stream>>>(xw2, deg, xgcn, n, OUTC);
  k_gcn_edge<<<blk((size_t)E * OUTC), T, 0, stream>>>(src, dst, xw2, deg, xgcn, E, OUTC);
  k_bias_act<<<blk((size_t)n * OUTC), T, 0, stream>>>(xgcn, b_gcn2, n, OUTC, 0);

  // P6-P9: GAT layer 1 (concat heads)
  k_gemm_bf16<IN_CH><<<dim3(GAT_HID / 16, n / 16), 32, 0, stream>>>(xbf, wp3, nullptr, xh1, n, GAT_HID);
  k_gat_alpha<<<blk((size_t)nH * 32), T, 0, stream>>>(xh1, a_src1, a_dst1, as_, ad_, n, NHEAD, HID);
  k_gat_max_init<<<blk(nH), T, 0, stream>>>(as_, ad_, mU, nH);
  k_gat_max_edge<<<blk((size_t)E * NHEAD), T, 0, stream>>>(src, dst, as_, ad_, mU, E, NHEAD);
  k_gat_max_dec<<<blk(nH), T, 0, stream>>>(mU, nH);
  k_gat_den_init<<<blk(nH), T, 0, stream>>>(as_, ad_, mF, den, nH);
  k_gat_out_init<<<blk((size_t)n * GAT_HID), T, 0, stream>>>(xh1, den, out1, n, NHEAD, HID);
  k_gat_edge<<<E, NHEAD * HID, 0, stream>>>(src, dst, as_, ad_, mF, xh1, out1, den, NHEAD, HID);
  k_gat_fin1<<<blk((size_t)n * GAT_HID), T, 0, stream>>>(out1, den, b_gat1, n, NHEAD, HID); // -> g1
  k_f32_to_bf16<<<blk((size_t)n * GAT_HID), T, 0, stream>>>(out1, g1bf, (size_t)n * GAT_HID);

  // P10-P11: GAT layer 2 (mean heads)
  k_gemm_bf16<GAT_HID><<<dim3(HOUT / 16, n / 16), 32, 0, stream>>>(g1bf, wp4, nullptr, xh2, n, HOUT);
  k_gat_alpha<<<blk((size_t)nH * 32), T, 0, stream>>>(xh2, a_src2, a_dst2, as_, ad_, n, NHEAD, OUTC);
  k_gat_max_init<<<blk(nH), T, 0, stream>>>(as_, ad_, mU, nH);
  k_gat_max_edge<<<blk((size_t)E * NHEAD), T, 0, stream>>>(src, dst, as_, ad_, mU, E, NHEAD);
  k_gat_max_dec<<<blk(nH), T, 0, stream>>>(mU, nH);
  k_gat_den_init<<<blk(nH), T, 0, stream>>>(as_, ad_, mF, den, nH);
  k_gat_out_init<<<blk((size_t)n * HOUT), T, 0, stream>>>(xh2, den, out2, n, NHEAD, OUTC);
  k_gat_edge<<<E, NHEAD * OUTC, 0, stream>>>(src, dst, as_, ad_, mF, xh2, out2, den, NHEAD, OUTC);
  k_gat_fin2<<<blk((size_t)n * OUTC), T, 0, stream>>>(out2, den, b_gat2, xgat, n, NHEAD, OUTC);

  // P12: LayerNorms -> bf16 concat buffer [n, 256]
  k_layernorm_bf16<<<blk((size_t)n * 32), T, 0, stream>>>(xgcn, g_gcn, bt_gcn, cat, n, OUTC, 2 * OUTC, 0);
  k_layernorm_bf16<<<blk((size_t)n * 32), T, 0, stream>>>(xgat, g_gat, bt_gat, cat, n, OUTC, 2 * OUTC, OUTC);

  // P13: fuse GEMM -> d_out (f32)
  k_gemm_bf16<2 * OUTC><<<dim3(OUTC / 16, n / 16), 32, 0, stream>>>(cat, wp5, b_fuse, (float*)d_out, n, OUTC);
}
